// MMTripletLoss_51728586113513
// MI455X (gfx1250) — compile-verified, matrix operand order checked by reading, not run on hardware
//
#include <hip/hip_runtime.h>
#include <hip/hip_bf16.h>
#include <cstdint>

typedef __attribute__((ext_vector_type(16))) __bf16 v16bf;
typedef __attribute__((ext_vector_type(8)))  __bf16 v8bf;
typedef __attribute__((ext_vector_type(8)))  float  v8f;
typedef __attribute__((ext_vector_type(4)))  float  v4f;
typedef __attribute__((ext_vector_type(4)))  unsigned int v4u;
typedef __attribute__((ext_vector_type(8)))  int    v8i;
typedef __attribute__((ext_vector_type(4)))  int    v4i;

#define NROWS 4096
#define DIMK  2048
#define BT    128          // 128x128 output tile per workgroup
#define KT    32           // K-slab per iteration (bf16 WMMA K)
#define KITER (DIMK / KT)  // 64
#define EPSD  1e-12f

// ---- order-preserving float<->uint encoding (for atomic max/min mining) ----
__device__ __forceinline__ unsigned encF(float f) {
  unsigned b = __float_as_uint(f);
  return (b & 0x80000000u) ? ~b : (b | 0x80000000u);
}
__device__ __forceinline__ float decF(unsigned k) {
  unsigned b = (k & 0x80000000u) ? (k ^ 0x80000000u) : ~k;
  return __uint_as_float(b);
}

// ---- TDM: load a [128 rows x 32 f32] tile of X (row stride DIMK) into LDS ----
__device__ __forceinline__ void tdm_load_tile(const float* gbase, int row0, int k0,
                                              const void* lds_ptr) {
  unsigned long long ga = (unsigned long long)(uintptr_t)gbase +
      ((unsigned long long)row0 * DIMK + (unsigned)k0) * 4ull;
  unsigned lds_off = (unsigned)(unsigned long long)(uintptr_t)lds_ptr;
  v4u g0;
  g0[0] = 1u;                                   // count=1, user descriptor
  g0[1] = lds_off;                              // LDS byte address
  g0[2] = (unsigned)(ga & 0xFFFFFFFFu);         // global_addr[31:0]
  g0[3] = (unsigned)((ga >> 32) & 0x1FFFFFFu)   // global_addr[56:32]
          | (2u << 30);                         // type=2 ("image")
  v8i g1;
  g1[0] = 0x20000;                              // wg_mask=0, data_size=2 (4B)
  g1[1] = (int)(DIMK << 16);                    // tensor_dim0 = 2048
  g1[2] = (int)(NROWS << 16);                   // tensor_dim1 = 4096
  g1[3] = (int)(KT << 16);                      // tile_dim0 = 32 (contiguous K)
  g1[4] = BT;                                   // tile_dim1 = 128 rows
  g1[5] = DIMK;                                 // tensor_dim0_stride = 2048
  g1[6] = 0;
  g1[7] = 0;
  v4i z4 = {0, 0, 0, 0};                        // 2D tensor: groups 2/3 unused
  v8i z8 = {0, 0, 0, 0, 0, 0, 0, 0};
  __builtin_amdgcn_tensor_load_to_lds(g0, g1, z4, z4, z8, 0);
}

// XOR swizzle of 16B chunks within a 64B row -> fragment reads hit 16 distinct
// bank groups (lanes 0..15 read consecutive rows at 64B stride).
__device__ __forceinline__ int chunk_sw(int row) {
  return (row & 3) ^ ((row >> 2) & 3);
}

// ---- convert one [128][32] f32 staged tile into swizzled bf16 hi/lo tiles.
// Each element split exactly once per workgroup: hi = bf16(f), lo = bf16(f-hi).
__device__ __forceinline__ void convert_tile(const float* st, __bf16* th,
                                             __bf16* tl, int tid) {
#pragma unroll
  for (int p = 0; p < 2; ++p) {
    const int L = p * 256 + tid;        // 8-element chunk id, 0..511
    const int row = L >> 2;
    const int c = L & 3;
    const float* src = st + row * KT + c * 8;
    v4f f0 = *(const v4f*)(src);
    v4f f1 = *(const v4f*)(src + 4);
    v8bf hi, lo;
#pragma unroll
    for (int e = 0; e < 4; ++e) {
      float a = f0[e];
      __bf16 h0 = (__bf16)a;
      hi[e] = h0;
      lo[e] = (__bf16)(a - (float)h0);
      float b = f1[e];
      __bf16 h1 = (__bf16)b;
      hi[4 + e] = h1;
      lo[4 + e] = (__bf16)(b - (float)h1);
    }
    const int dst = row * KT + ((c ^ chunk_sw(row)) * 8);
    *(v8bf*)(th + dst) = hi;
    *(v8bf*)(tl + dst) = lo;
  }
}

// ---- load a 16x32 bf16 fragment: two contiguous 16B runs (2x ds_load_b128).
// ISA 16-bit A layout: lanes 0-15 -> M=lane, K {0..7,16..23};
// lanes 16-31 -> M=lane-16, K {8..15,24..31}. Same pattern assumed for B (NxK).
__device__ __forceinline__ v16bf load_frag_bf(const __bf16* t, int row, int half) {
  const int sw = chunk_sw(row);
  const int c1 = half ^ sw;        // K = half*8 .. half*8+7
  const int c2 = (2 + half) ^ sw;  // K = 16+half*8 .. +7
  v8bf q0 = *(const v8bf*)(t + row * KT + c1 * 8);
  v8bf q1 = *(const v8bf*)(t + row * KT + c2 * 8);
  return __builtin_shufflevector(q0, q1, 0, 1, 2, 3, 4, 5, 6, 7,
                                 8, 9, 10, 11, 12, 13, 14, 15);
}

// ================= kernel 1: row squared-norms + init mining arrays =========
__global__ __launch_bounds__(256)
void sqnorm_init_kernel(const float* __restrict__ R, const float* __restrict__ Nm,
                        const float* __restrict__ T, float* __restrict__ sq,
                        unsigned* __restrict__ apE, unsigned* __restrict__ anE) {
  const int b = blockIdx.x;
  const int ii = b >> 12;          // input index 0..2
  const int row = b & (NROWS - 1);
  const float* X = (ii == 0) ? R : (ii == 1) ? Nm : T;
  const float* xr = X + (size_t)row * DIMK;
  float s = 0.0f;
  for (int k = threadIdx.x; k < DIMK; k += 256) {
    float v = xr[k];
    s += v * v;
  }
#pragma unroll
  for (int m = 1; m <= 16; m <<= 1) s += __shfl_xor(s, m, 32);
  __shared__ float wsum[8];
  if ((threadIdx.x & 31) == 0) wsum[threadIdx.x >> 5] = s;
  __syncthreads();
  if (threadIdx.x == 0) {
    float tot = 0.0f;
#pragma unroll
    for (int w = 0; w < 8; ++w) tot += wsum[w];
    const int idx = ii * NROWS + row;
    sq[idx] = tot;
    apE[idx] = 0u;           // < enc of any real value (max-mining init)
    anE[idx] = 0xFFFFFFFFu;  // > enc of any real value (min-mining init)
  }
}

// ===== kernel 2: TDM-fed, LDS-split bf16x3 Gram tile + hardest mining =======
__global__ __launch_bounds__(256)
void hardest_tile_kernel(const float* __restrict__ R, const float* __restrict__ Nm,
                         const float* __restrict__ T, const int* __restrict__ tgt,
                         const float* __restrict__ sq, unsigned* __restrict__ apE,
                         unsigned* __restrict__ anE) {
  __shared__ __align__(16) float  stA[BT * KT];   // 16KB TDM staging (f32)
  __shared__ __align__(16) float  stB[BT * KT];   // 16KB
  __shared__ __align__(16) __bf16 Ah[BT * KT];    // 8KB swizzled bf16 hi
  __shared__ __align__(16) __bf16 Al[BT * KT];    // 8KB swizzled bf16 lo
  __shared__ __align__(16) __bf16 Bh[BT * KT];    // 8KB
  __shared__ __align__(16) __bf16 Bl[BT * KT];    // 8KB  -> 64KB total

  const int cb = blockIdx.x, rb = blockIdx.y, ii = blockIdx.z;
  if (cb < rb) return;               // symmetric: upper-triangular tiles only
  const float* X = (ii == 0) ? R : (ii == 1) ? Nm : T;
  const float* sqi = sq + ii * NROWS;
  unsigned* apI = apE + ii * NROWS;
  unsigned* anI = anE + ii * NROWS;

  const int tid  = threadIdx.x;
  const int wave = tid >> 5;
  const int lane = tid & 31;
  const int half = lane >> 4;        // C/D: lanes 16-31 hold M+8
  const int l15  = lane & 15;
  const int waveRow = wave >> 1;     // 0..3 -> 32-row strip
  const int waveCol = wave & 1;      // 0..1 -> 64-col strip
  const int rowW = rb * BT + waveRow * 32;
  const int colW = cb * BT + waveCol * 64;

  v8f acc[2][4];
#pragma unroll
  for (int a = 0; a < 2; ++a)
#pragma unroll
    for (int c = 0; c < 4; ++c)
#pragma unroll
      for (int e = 0; e < 8; ++e) acc[a][c][e] = 0.0f;

  // prologue: wave0 -> A slab, wave1 -> B slab (TDM, TENSORcnt-tracked)
  if (wave == 0)      tdm_load_tile(X, rb * BT, 0, stA);
  else if (wave == 1) tdm_load_tile(X, cb * BT, 0, stB);

  for (int kt = 0; kt < KITER; ++kt) {
    __builtin_amdgcn_s_wait_tensorcnt(0);  // no-op for waves with no TDM ops
    __syncthreads();   // slab kt staged; previous iteration's compute finished

    convert_tile(stA, Ah, Al, tid);        // split each element exactly once
    convert_tile(stB, Bh, Bl, tid);
    __syncthreads();   // bf16 tiles ready; f32 stage is free again

    if (kt + 1 < KITER) {                  // TDM prefetch overlaps compute
      if (wave == 0)      tdm_load_tile(X, rb * BT, (kt + 1) * KT, stA);
      else if (wave == 1) tdm_load_tile(X, cb * BT, (kt + 1) * KT, stB);
    }

    v16bf ah[2], al[2];
#pragma unroll
    for (int rt = 0; rt < 2; ++rt) {
      const int ar = waveRow * 32 + rt * 16 + l15;
      ah[rt] = load_frag_bf(Ah, ar, half);
      al[rt] = load_frag_bf(Al, ar, half);
    }
#pragma unroll
    for (int ct = 0; ct < 4; ++ct) {
      const int br = waveCol * 64 + ct * 16 + l15;
      v16bf bh = load_frag_bf(Bh, br, half);
      v16bf bl = load_frag_bf(Bl, br, half);
#pragma unroll
      for (int rt = 0; rt < 2; ++rt) {
        // fp32 emulation: hi*hi + hi*lo + lo*hi into one f32 accumulator
        acc[rt][ct] = __builtin_amdgcn_wmma_f32_16x16x32_bf16(
            false, ah[rt], false, bh, (short)0, acc[rt][ct], false, false);
        acc[rt][ct] = __builtin_amdgcn_wmma_f32_16x16x32_bf16(
            false, ah[rt], false, bl, (short)0, acc[rt][ct], false, false);
        acc[rt][ct] = __builtin_amdgcn_wmma_f32_16x16x32_bf16(
            false, al[rt], false, bh, (short)0, acc[rt][ct], false, false);
      }
    }
  }

  // ---------------- epilogue: distances + hardest mining -------------------
  float sqR[2][8]; int tR[2][8];
#pragma unroll
  for (int rt = 0; rt < 2; ++rt)
#pragma unroll
    for (int r = 0; r < 8; ++r) {
      const int gr = rowW + rt * 16 + half * 8 + r;
      sqR[rt][r] = sqi[gr];
      tR[rt][r] = tgt[gr];
    }
  float sqC[4]; int tC[4];
#pragma unroll
  for (int ct = 0; ct < 4; ++ct) {
    const int gc = colW + ct * 16 + l15;
    sqC[ct] = sqi[gc];
    tC[ct] = tgt[gc];
  }
  // acc -> sqrt(clamp(||xi||^2 + ||xj||^2 - 2 xi.xj, eps))
#pragma unroll
  for (int rt = 0; rt < 2; ++rt)
#pragma unroll
    for (int ct = 0; ct < 4; ++ct)
#pragma unroll
      for (int r = 0; r < 8; ++r) {
        float d2 = sqR[rt][r] + sqC[ct] - 2.0f * acc[rt][ct][r];
        acc[rt][ct][r] = sqrtf(fmaxf(d2, EPSD));
      }

  const float INF = __builtin_inff();
  // row-wise: reduce over the 64 columns this wave owns
#pragma unroll
  for (int rt = 0; rt < 2; ++rt)
#pragma unroll
    for (int r = 0; r < 8; ++r) {
      float ap = -INF, an = INF;
      const int tr = tR[rt][r];
#pragma unroll
      for (int ct = 0; ct < 4; ++ct) {
        float d = acc[rt][ct][r];
        if (tr == tC[ct]) ap = fmaxf(ap, d);
        else              an = fminf(an, d);
      }
#pragma unroll
      for (int m = 1; m <= 8; m <<= 1) {   // reduce over N within lane-half
        ap = fmaxf(ap, __shfl_xor(ap, m, 32));
        an = fminf(an, __shfl_xor(an, m, 32));
      }
      if (l15 == 0) {
        const int gr = rowW + rt * 16 + half * 8 + r;
        atomicMax(&apI[gr], encF(ap));
        atomicMin(&anI[gr], encF(an));
      }
    }
  // col-wise: covers the transposed (j,i) entries of off-diagonal tiles
#pragma unroll
  for (int ct = 0; ct < 4; ++ct) {
    float ap = -INF, an = INF;
    const int tc = tC[ct];
#pragma unroll
    for (int rt = 0; rt < 2; ++rt)
#pragma unroll
      for (int r = 0; r < 8; ++r) {
        float d = acc[rt][ct][r];
        if (tR[rt][r] == tc) ap = fmaxf(ap, d);
        else                 an = fminf(an, d);
      }
    ap = fmaxf(ap, __shfl_xor(ap, 16, 32));  // join the two M-halves
    an = fminf(an, __shfl_xor(an, 16, 32));
    if (half == 0) {
      const int gc = colW + ct * 16 + l15;
      atomicMax(&apI[gc], encF(ap));
      atomicMin(&anI[gc], encF(an));
    }
  }
}

// ============ kernel 3: combine 3 inputs, margin ranking loss mean ==========
__global__ __launch_bounds__(256)
void final_loss_kernel(const unsigned* __restrict__ apE,
                       const unsigned* __restrict__ anE,
                       float* __restrict__ out) {
  float s = 0.0f;
  for (int r = threadIdx.x; r < NROWS; r += 256) {
    float ap = fminf(fminf(decF(apE[r]), decF(apE[NROWS + r])), decF(apE[2 * NROWS + r]));
    float an = fmaxf(fmaxf(decF(anE[r]), decF(anE[NROWS + r])), decF(anE[2 * NROWS + r]));
    s += fmaxf(0.0f, ap - an + 1.0f);
  }
#pragma unroll
  for (int m = 1; m <= 16; m <<= 1) s += __shfl_xor(s, m, 32);
  __shared__ float wsum[8];
  if ((threadIdx.x & 31) == 0) wsum[threadIdx.x >> 5] = s;
  __syncthreads();
  if (threadIdx.x == 0) {
    float tot = 0.0f;
#pragma unroll
    for (int w = 0; w < 8; ++w) tot += wsum[w];
    out[0] = tot / (float)NROWS;
  }
}

extern "C" void kernel_launch(void* const* d_in, const int* in_sizes, int n_in,
                              void* d_out, int out_size, void* d_ws, size_t ws_size,
                              hipStream_t stream) {
  const float* R  = (const float*)d_in[0];
  const float* Nm = (const float*)d_in[1];
  const float* T  = (const float*)d_in[2];
  const int* tgt  = (const int*)d_in[3];
  float* sq       = (float*)d_ws;
  unsigned* apE   = (unsigned*)((char*)d_ws + (size_t)3 * NROWS * sizeof(float));
  unsigned* anE   = apE + 3 * NROWS;

  sqnorm_init_kernel<<<3 * NROWS, 256, 0, stream>>>(R, Nm, T, sq, apE, anE);
  dim3 grid(NROWS / BT, NROWS / BT, 3);   // 32 x 32 x 3, cb<rb exit early
  hardest_tile_kernel<<<grid, 256, 0, stream>>>(R, Nm, T, tgt, sq, apE, anE);
  final_loss_kernel<<<1, 256, 0, stream>>>(apE, anE, (float*)d_out);
}
